// GAT_6055903887406
// MI455X (gfx1250) — compile-verified, matrix operand order checked by reading, not run on hardware
//
#include <hip/hip_runtime.h>
#include <hip/hip_bf16.h>
#include <math.h>

// ---------------------------------------------------------------------------
// Types for CDNA5 WMMA (wave32, 16x16x32 bf16 -> f32)
// ---------------------------------------------------------------------------
typedef __bf16 bf16_t;
typedef bf16_t bf16x16 __attribute__((ext_vector_type(16)));
typedef bf16_t bf16x8  __attribute__((ext_vector_type(8)));
typedef float  v8f     __attribute__((ext_vector_type(8)));

union Frag16 { bf16x16 v; bf16x8 h[2]; };

__device__ __forceinline__ void atomicMaxF32(float* addr, float val) {
  // Monotonic bit-trick: non-negative floats compare as signed ints,
  // negative floats compare (reversed) as unsigned ints. Correct with -inf init.
  if (val >= 0.0f) atomicMax((int*)addr, __float_as_int(val));
  else             atomicMin((unsigned int*)addr, __float_as_uint(val));
}

// ---------------------------------------------------------------------------
// Elementwise helpers
// ---------------------------------------------------------------------------
__global__ void fill_f32(float* __restrict__ p, float v, long n) {
  long i = (long)blockIdx.x * blockDim.x + threadIdx.x;
  if (i < n) p[i] = v;
}

__global__ void cvt_bf16(const float* __restrict__ s, bf16_t* __restrict__ d, long n) {
  long i = (long)blockIdx.x * blockDim.x + threadIdx.x;
  if (i < n) d[i] = (bf16_t)s[i];
}

// ---------------------------------------------------------------------------
// Pack fp32 weight W[K x Nreal] (row major) into bf16 B-fragment layout.
// Fragment (kt,nt) = 32 lanes x 16 bf16:
//   lane L (<16):  N = nt*16+L,      K = kt*32 + {0..7, 16..23}
//   lane L (>=16): N = nt*16+(L-16), K = kt*32 + {8..15, 24..31}
// Zero-pad N beyond Nreal.
// ---------------------------------------------------------------------------
__global__ void pack_w(const float* __restrict__ W, bf16_t* __restrict__ Bp,
                       int K, int Nreal, int nTilesN) {
  int total = (K >> 5) * nTilesN * 512;
  int i = blockIdx.x * blockDim.x + threadIdx.x;
  if (i >= total) return;
  int j    = i & 15;
  int lane = (i >> 4) & 31;
  int frag = i >> 9;
  int nt = frag % nTilesN;
  int kt = frag / nTilesN;
  int ncol = nt * 16 + (lane & 15);
  int klo  = (lane & 16) ? 8 : 0;
  int k = kt * 32 + klo + (j & 7) + ((j >> 3) << 4);
  float v = (ncol < Nreal) ? W[(size_t)k * Nreal + ncol] : 0.0f;
  Bp[i] = (bf16_t)v;
}

// ---------------------------------------------------------------------------
// WMMA GEMM: C[n x Nreal] = A[n x K] * W  (A bf16 row-major, W pre-packed bf16)
// n % 16 == 0, K % 32 == 0, nTilesN == nGroups * NT (exact).
// One wave computes a 16 x (NT*16) strip. NT is compile-time so the inner
// loop is straight-line. All fragment loads for a k-step are issued before
// the WMMAs so the compiler can use partial s_wait_loadcnt and overlap
// load latency with the matrix pipe.
// ---------------------------------------------------------------------------
template <int NT, bool GUARD>
__global__ void wmma_gemm_t(const bf16_t* __restrict__ A, const bf16_t* __restrict__ Bp,
                            float* __restrict__ C, int n, int K, int Nreal,
                            int nTilesN, int nGroups) {
  const int lane = threadIdx.x & 31;
  const int totalW = (n >> 4) * nGroups;
  const int wid = blockIdx.x * (blockDim.x >> 5) + (threadIdx.x >> 5);
  if (wid >= totalW) return;  // wave-uniform exit: EXEC all-ones below

  const int mt  = wid / nGroups;
  const int nt0 = (wid % nGroups) * NT;
  const int klo = (lane & 16) ? 8 : 0;

  const bf16_t* __restrict__ aptr =
      A + (size_t)(mt * 16 + (lane & 15)) * K + klo;
  const bf16_t* __restrict__ bptr =
      Bp + (((size_t)nt0 << 5) + lane) * 16;
  const size_t bstep = (size_t)nTilesN << 9;   // fragments per k-step * 512

  v8f acc[NT];
#pragma unroll
  for (int j = 0; j < NT; ++j) acc[j] = v8f{0.f, 0.f, 0.f, 0.f, 0.f, 0.f, 0.f, 0.f};

  const int KT = K >> 5;
  for (int kt = 0; kt < KT; ++kt) {
    // Issue all loads for this k-step first.
    Frag16 a;
    a.h[0] = *(const bf16x8*)(aptr);
    a.h[1] = *(const bf16x8*)(aptr + 16);
    Frag16 b[NT];
#pragma unroll
    for (int j = 0; j < NT; ++j) {
      b[j].h[0] = *(const bf16x8*)(bptr + j * 512);
      b[j].h[1] = *(const bf16x8*)(bptr + j * 512 + 8);
    }
    // Then the matrix ops.
#pragma unroll
    for (int j = 0; j < NT; ++j) {
      acc[j] = __builtin_amdgcn_wmma_f32_16x16x32_bf16(
          false, a.v, false, b[j].v, (short)0, acc[j], false, false);
    }
    aptr += 32;
    bptr += bstep;
  }

  // C/D layout: VGPR r -> row (r + 8*(lane>=16)), col = lane&15
  const int rbase = (lane & 16) ? 8 : 0;
  const int col   = lane & 15;
#pragma unroll
  for (int j = 0; j < NT; ++j) {
    const int nc = (nt0 + j) * 16 + col;
    float* crow = C + (size_t)(mt * 16 + rbase) * Nreal + nc;
    if (!GUARD || nc < Nreal) {
#pragma unroll
      for (int r = 0; r < 8; ++r) crow[(size_t)r * Nreal] = acc[j][r];
    }
  }
}

// ---------------------------------------------------------------------------
// Attention coefficients: out[n,h] = sum_d fs[n, h*D+d] * a[h*D+d]
// ---------------------------------------------------------------------------
__global__ void attn_coef(const float* __restrict__ fs, const float* __restrict__ avec,
                          float* __restrict__ out, int n, int D, int HD) {
  long i = (long)blockIdx.x * blockDim.x + threadIdx.x;
  if (i >= (long)n * 4) return;
  long node = i >> 2;
  int  h = (int)(i & 3);
  const float* f = fs + node * HD + h * D;
  const float* a = avec + h * D;
  float s = 0.f;
  for (int d = 0; d < D; ++d) s += f[d] * a[d];
  out[i] = s;
}

// ---------------------------------------------------------------------------
// Edge pass 1: e = leaky_relu(el[src]+er[dst], 0.2); store e; segment max.
// ---------------------------------------------------------------------------
__global__ void edge_score(const int* __restrict__ src, const int* __restrict__ dst,
                           const float* __restrict__ el, const float* __restrict__ er,
                           float* __restrict__ eb, float* __restrict__ emax, long ne) {
  long i = (long)blockIdx.x * blockDim.x + threadIdx.x;
  if (i >= ne * 4) return;
  long e = i >> 2;
  int  h = (int)(i & 3);
  int  s  = src[e];
  int  dt = dst[e];
  float v = el[(long)s * 4 + h] + er[(long)dt * 4 + h];
  v = (v > 0.f) ? v : 0.2f * v;
  eb[i] = v;
  atomicMaxF32(&emax[(long)dt * 4 + h], v);
}

// Edge pass 2: ee = exp(e - emax[dst]); denom[dst] += ee.
__global__ void edge_exp(const int* __restrict__ dst, float* __restrict__ eb,
                         const float* __restrict__ emax, float* __restrict__ den, long ne) {
  long i = (long)blockIdx.x * blockDim.x + threadIdx.x;
  if (i >= ne * 4) return;
  long e = i >> 2;
  int  h = (int)(i & 3);
  int  dt = dst[e];
  float w = __expf(eb[i] - emax[(long)dt * 4 + h]);
  eb[i] = w;
  atomicAdd(&den[(long)dt * 4 + h], w);
}

// Edge pass 3 (layers 0/1, HD=256, D=64): float4 chunks, atomicAdd scatter.
__global__ void edge_msg_v4(const int* __restrict__ src, const int* __restrict__ dst,
                            const float* __restrict__ fs, const float* __restrict__ ee,
                            const float* __restrict__ den, float* __restrict__ rst, long ne) {
  long i = (long)blockIdx.x * blockDim.x + threadIdx.x;
  if (i >= ne * 64) return;
  long e  = i >> 6;
  int  c4 = (int)(i & 63) << 2;
  int  h  = c4 >> 6;               // D = 64
  int  s  = src[e];
  int  dt = dst[e];
  float alpha = ee[e * 4 + h] / den[(long)dt * 4 + h];
  const float4 f = *(const float4*)(fs + (long)s * 256 + c4);
  float* o = rst + (long)dt * 256 + c4;
  atomicAdd(o + 0, alpha * f.x);
  atomicAdd(o + 1, alpha * f.y);
  atomicAdd(o + 2, alpha * f.z);
  atomicAdd(o + 3, alpha * f.w);
}

// Edge pass 3 (layer 2, HD=188, D=47): scalar.
__global__ void edge_msg_l2(const int* __restrict__ src, const int* __restrict__ dst,
                            const float* __restrict__ fs, const float* __restrict__ ee,
                            const float* __restrict__ den, float* __restrict__ rst, long ne) {
  long i = (long)blockIdx.x * blockDim.x + threadIdx.x;
  if (i >= ne * 188) return;
  long e = i / 188;
  int  c = (int)(i - e * 188);
  int  h = c / 47;
  int  s  = src[e];
  int  dt = dst[e];
  float alpha = ee[e * 4 + h] / den[(long)dt * 4 + h];
  atomicAdd(&rst[(long)dt * 188 + c], alpha * fs[(long)s * 188 + c]);
}

// Layers 0/1 epilogue: h = elu(rst + b + skip + bs), in place. HD == 256.
__global__ void post_elu(float* __restrict__ rst, const float* __restrict__ b,
                         const float* __restrict__ skip, const float* __restrict__ bs,
                         long n) {
  long i = (long)blockIdx.x * blockDim.x + threadIdx.x;
  if (i >= n) return;
  int c = (int)(i & 255);
  float v = rst[i] + b[c] + skip[i] + bs[c];
  rst[i] = (v > 0.f) ? v : (__expf(v) - 1.0f);
}

// Layer 2 epilogue: head-mean + skip + bias, then log_softmax over 47 classes.
// One wave (32 lanes) per row; lane handles cols j and j+32.
__global__ void final_lsm(const float* __restrict__ rst, const float* __restrict__ b2,
                          const float* __restrict__ skip, const float* __restrict__ bs2,
                          float* __restrict__ out) {
  const int C = 47, H = 4;
  int node = blockIdx.x;
  int lane = threadIdx.x;
  float t0 = -1e30f, t1 = -1e30f;
#pragma unroll
  for (int q = 0; q < 2; ++q) {
    int j = lane + q * 32;
    if (j < C) {
      float s = 0.f;
#pragma unroll
      for (int h = 0; h < H; ++h) s += rst[(long)node * 188 + h * C + j] + b2[h * C + j];
      float v = s * 0.25f + skip[(long)node * C + j] + bs2[j];
      if (q == 0) t0 = v; else t1 = v;
    }
  }
  float mx = fmaxf(t0, t1);
  for (int o = 16; o > 0; o >>= 1) mx = fmaxf(mx, __shfl_xor(mx, o, 32));
  float se = 0.f;
  if (lane < C)      se += __expf(t0 - mx);
  if (lane + 32 < C) se += __expf(t1 - mx);
  for (int o = 16; o > 0; o >>= 1) se += __shfl_xor(se, o, 32);
  float lse = __logf(se);
  if (lane < C)      out[(long)node * C + lane]      = t0 - mx - lse;
  if (lane + 32 < C) out[(long)node * C + lane + 32] = t1 - mx - lse;
}

// ---------------------------------------------------------------------------
// Host launcher
// ---------------------------------------------------------------------------
static inline unsigned cdiv(long a, long b) { return (unsigned)((a + b - 1) / b); }

static void launch_gemm(const bf16_t* A, const bf16_t* Bp, float* C,
                        int n, int K, int Nreal, int nTilesN, hipStream_t stream) {
  if (nTilesN % 4 == 0) {
    const int nGroups = nTilesN / 4;
    const long waves = (long)(n / 16) * nGroups;
    if (Nreal % 16 == 0) {
      wmma_gemm_t<4, false><<<cdiv(waves, 8), 256, 0, stream>>>(
          A, Bp, C, n, K, Nreal, nTilesN, nGroups);
    } else {
      wmma_gemm_t<4, true><<<cdiv(waves, 8), 256, 0, stream>>>(
          A, Bp, C, n, K, Nreal, nTilesN, nGroups);
    }
  } else {  // nTilesN == 3 (layer-2 skip GEMM, Nreal = 47)
    const int nGroups = 1;
    const long waves = (long)(n / 16) * nGroups;
    wmma_gemm_t<3, true><<<cdiv(waves, 8), 256, 0, stream>>>(
        A, Bp, C, n, K, Nreal, nTilesN, nGroups);
  }
}

extern "C" void kernel_launch(void* const* d_in, const int* in_sizes, int n_in,
                              void* d_out, int out_size, void* d_ws, size_t ws_size,
                              hipStream_t stream) {
  (void)in_sizes; (void)n_in; (void)out_size; (void)ws_size;

  const float* x = (const float*)d_in[0];
  const float* W[3]  = {(const float*)d_in[1],  (const float*)d_in[7],  (const float*)d_in[13]};
  const float* al[3] = {(const float*)d_in[2],  (const float*)d_in[8],  (const float*)d_in[14]};
  const float* ar[3] = {(const float*)d_in[3],  (const float*)d_in[9],  (const float*)d_in[15]};
  const float* bb[3] = {(const float*)d_in[4],  (const float*)d_in[10], (const float*)d_in[16]};
  const float* Wsk[3]= {(const float*)d_in[5],  (const float*)d_in[11], (const float*)d_in[17]};
  const float* bs[3] = {(const float*)d_in[6],  (const float*)d_in[12], (const float*)d_in[18]};
  const int* srcs[3] = {(const int*)d_in[19], (const int*)d_in[21], (const int*)d_in[23]};
  const int* dsts[3] = {(const int*)d_in[20], (const int*)d_in[22], (const int*)d_in[24]};

  const int  nsrc[3] = {200000, 50000, 10000};
  const int  ndst[3] = {50000, 10000, 2048};
  const int  cin[3]  = {128, 256, 256};
  const int  Dh[3]   = {64, 64, 47};
  const int  HD[3]   = {256, 256, 188};
  const int  soutL[3]= {256, 256, 47};
  const long ne[3]   = {800000, 400000, 100000};

  // ---- workspace arena (bump allocator, 256B aligned) ----
  size_t off = 0;
  auto arena = [&](size_t bytes) -> void* {
    void* p = (char*)d_ws + off;
    off += (bytes + 255) & ~(size_t)255;
    return p;
  };
  bf16_t* hbf  = (bf16_t*)arena(200000ull * 128 * 2);   // activations in bf16 (max layer)
  float*  fs   = (float*) arena(200000ull * 256 * 4);   // conv features
  float*  skp  = (float*) arena(50000ull  * 256 * 4);   // skip-GEMM output
  float*  el   = (float*) arena(200000ull * 4 * 4);
  float*  er   = (float*) arena(50000ull  * 4 * 4);
  float*  emax = (float*) arena(50000ull  * 4 * 4);
  float*  den  = (float*) arena(50000ull  * 4 * 4);
  float*  ee   = (float*) arena(800000ull * 4 * 4);     // edge scores / exp
  float*  hA   = (float*) arena(50000ull  * 256 * 4);   // activation ping
  float*  hB   = (float*) arena(10000ull  * 256 * 4);   // activation pong
  bf16_t* Wp   = (bf16_t*)arena(256ull * 256 * 2);      // packed conv weight
  bf16_t* Wsp  = (bf16_t*)arena(256ull * 256 * 2);      // packed skip weight

  const float* hin = x;
  float* rstBuf[3] = {hA, hB, hA};

  for (int l = 0; l < 3; ++l) {
    const int n  = nsrc[l], nd = ndst[l], K = cin[l];
    const int D  = Dh[l],  hd = HD[l],  so = soutL[l];
    const int nTc = (hd + 15) / 16;     // conv N tiles (padded)
    const int nTs = (so + 15) / 16;     // skip N tiles (padded)

    // 1) fp32 -> bf16 activations
    long nIn = (long)n * K;
    cvt_bf16<<<cdiv(nIn, 256), 256, 0, stream>>>(hin, hbf, nIn);

    // 2) pack weights into WMMA B-fragment layout
    pack_w<<<cdiv((long)(K / 32) * nTc * 512, 256), 256, 0, stream>>>(W[l],   Wp,  K, hd, nTc);
    pack_w<<<cdiv((long)(K / 32) * nTs * 512, 256), 256, 0, stream>>>(Wsk[l], Wsp, K, so, nTs);

    // 3) conv GEMM: fs = hbf @ W   (n x hd)
    launch_gemm(hbf, Wp, fs, n, K, hd, nTc, stream);
    // 4) skip GEMM: skp = hbf[:nd] @ Ws   (nd x so); h_dst is a prefix of h_src
    launch_gemm(hbf, Wsp, skp, nd, K, so, nTs, stream);

    // 5) attention coefficients (fd = fs[:nd])
    attn_coef<<<cdiv((long)n * 4, 256), 256, 0, stream>>>(fs, al[l], el, n, D, hd);
    attn_coef<<<cdiv((long)nd * 4, 256), 256, 0, stream>>>(fs, ar[l], er, nd, D, hd);

    // 6) init segment buffers + output accumulator
    long nh = (long)nd * 4;
    fill_f32<<<cdiv(nh, 256), 256, 0, stream>>>(emax, -INFINITY, nh);
    fill_f32<<<cdiv(nh, 256), 256, 0, stream>>>(den, 0.0f, nh);
    float* rst = rstBuf[l];
    long nout = (long)nd * hd;
    fill_f32<<<cdiv(nout, 256), 256, 0, stream>>>(rst, 0.0f, nout);

    // 7) edge softmax + message aggregation
    long neh = ne[l] * 4;
    edge_score<<<cdiv(neh, 256), 256, 0, stream>>>(srcs[l], dsts[l], el, er, ee, emax, ne[l]);
    edge_exp<<<cdiv(neh, 256), 256, 0, stream>>>(dsts[l], ee, emax, den, ne[l]);

    if (l < 2) {
      edge_msg_v4<<<cdiv(ne[l] * 64, 256), 256, 0, stream>>>(srcs[l], dsts[l], fs, ee, den, rst, ne[l]);
      post_elu<<<cdiv(nout, 256), 256, 0, stream>>>(rst, bb[l], skp, bs[l], nout);
      hin = rst;
    } else {
      edge_msg_l2<<<cdiv(ne[l] * 188, 256), 256, 0, stream>>>(srcs[l], dsts[l], fs, ee, den, rst, ne[l]);
      final_lsm<<<dim3(2048), dim3(32), 0, stream>>>(rst, bb[l], skp, bs[l], (float*)d_out);
    }
  }
}
